// MHAttRela_49366354100331
// MI455X (gfx1250) — matrix operand order repaired and checked by reading, not compiled, not run on hardware
//
#include <hip/hip_runtime.h>

// ---------------------------------------------------------------------------
// Types for CDNA5 WMMA (wave32, 16x16x32 bf16 -> f32)
// ---------------------------------------------------------------------------
typedef __bf16 bf16_t;
typedef bf16_t v16bf __attribute__((ext_vector_type(16)));
typedef bf16_t v8bf  __attribute__((ext_vector_type(8)));
typedef float  v8f   __attribute__((ext_vector_type(8)));

static __device__ inline v8f wmma_bf16(v16bf a, v16bf b, v8f c) {
    // (neg_a, A, neg_b, B, c_mod, C, reuse_a, reuse_b)
    return __builtin_amdgcn_wmma_f32_16x16x32_bf16(false, a, false, b,
                                                   (short)0, c, false, false);
}

// A fragment, 16-bit 16x32 (M x K), same k order as B fragment:
//   lanes 0-15 : row M=lane,   K = k0..k0+15
//   lanes16-31 : row M=lane-16,K = k0+16..k0+31
// => lane holds 16 contiguous elements at k0+16*half.
static __device__ inline v16bf load_a_frag(const bf16_t* __restrict__ base,
                                           int ld, int m0, int k0, int lane) {
    int row  = m0 + (lane & 15);
    int half = lane >> 4;
    const bf16_t* p = base + (size_t)row * ld + k0 + half * 16;
    v8bf lo = *(const v8bf*)p;
    v8bf hi = *(const v8bf*)(p + 8);
    v16bf r;
    #pragma unroll
    for (int i = 0; i < 8; ++i) { r[i] = lo[i]; r[8 + i] = hi[i]; }
    return r;
}

// B fragment, 16-bit 32x16 (K x N), with B = Wt and W stored row-major (N,K):
//   lanes 0-15 : col N=lane,    K = k0..k0+15
//   lanes16-31 : col N=lane-16, K = k0+16..k0+31
// => one 32B contiguous load from row (n0+lane&15) of W.
static __device__ inline v16bf load_b_frag(const bf16_t* __restrict__ base,
                                           int ld, int n0, int k0, int lane) {
    int col  = n0 + (lane & 15);
    int half = lane >> 4;
    const bf16_t* p = base + (size_t)col * ld + k0 + half * 16;
    return *(const v16bf*)p;
}

// ---------------------------------------------------------------------------
// Problem constants
// ---------------------------------------------------------------------------
#define BB   8
#define SS   1024
#define HH   1024
#define NHH  16
#define DHH  64
#define MM   (BB * SS)   // 8192

// ---------------------------------------------------------------------------
// fp32 -> bf16 convert
// ---------------------------------------------------------------------------
__global__ void cvt_f32_bf16(const float* __restrict__ in,
                             bf16_t* __restrict__ out, int n) {
    int i = blockIdx.x * blockDim.x + threadIdx.x;
    int stride = gridDim.x * blockDim.x;
    for (; i < n; i += stride) out[i] = (bf16_t)in[i];
}

// ---------------------------------------------------------------------------
// GEMM: Y[m,n] = sum_k A[m,k]*W[n,k] (+ A2[m,k]*W2[n,k]) + bias[n] (+bias2[n])
// A: (8192 x 1024) bf16 row-major, W: (1024 x 1024) bf16 row-major.
// Block = 8 waves (256 thr), 128x128 tile; each wave a 32x64 tile
// (2x4 WMMA accs -> 8 WMMA per K-step on 6 fragment loads, ~22 FLOP/B vs L2).
// mode 0: bf16 head layout [B,NH,S,DH]
// mode 1: bf16 transposed head layout [B,NH,DH,S]   (for V)
// mode 2: f32 plain [M,N]                            (final output)
// ---------------------------------------------------------------------------
__global__ __launch_bounds__(256) void gemm_bf16(
    const bf16_t* __restrict__ A, const bf16_t* __restrict__ A2,
    const bf16_t* __restrict__ W, const bf16_t* __restrict__ W2,
    const float* __restrict__ bias, const float* __restrict__ bias2,
    void* __restrict__ out, int mode)
{
    const int K = HH;
    int lane = threadIdx.x & 31;
    int wave = threadIdx.x >> 5;
    int wm = wave & 3, wn = wave >> 2;
    int m0 = blockIdx.y * 128 + wm * 32;
    int n0 = blockIdx.x * 128 + wn * 64;

    v8f acc[2][4] = {};

    const int npass = A2 ? 2 : 1;
    for (int pass = 0; pass < npass; ++pass) {
        const bf16_t* Ap = pass ? A2 : A;
        const bf16_t* Wp = pass ? W2 : W;
        for (int kk = 0; kk < K; kk += 32) {
            if (kk + 32 < K) {   // warm next K-tile into cache
                __builtin_prefetch(Ap + (size_t)(m0 + (lane & 15)) * K + kk + 32, 0, 3);
                __builtin_prefetch(Wp + (size_t)(n0 + (lane & 15)) * K + kk + 32, 0, 3);
            }
            v16bf a0 = load_a_frag(Ap, K, m0,      kk, lane);
            v16bf a1 = load_a_frag(Ap, K, m0 + 16, kk, lane);
            v16bf b0 = load_b_frag(Wp, K, n0,      kk, lane);
            v16bf b1 = load_b_frag(Wp, K, n0 + 16, kk, lane);
            v16bf b2 = load_b_frag(Wp, K, n0 + 32, kk, lane);
            v16bf b3 = load_b_frag(Wp, K, n0 + 48, kk, lane);
            acc[0][0] = wmma_bf16(a0, b0, acc[0][0]);
            acc[0][1] = wmma_bf16(a0, b1, acc[0][1]);
            acc[0][2] = wmma_bf16(a0, b2, acc[0][2]);
            acc[0][3] = wmma_bf16(a0, b3, acc[0][3]);
            acc[1][0] = wmma_bf16(a1, b0, acc[1][0]);
            acc[1][1] = wmma_bf16(a1, b1, acc[1][1]);
            acc[1][2] = wmma_bf16(a1, b2, acc[1][2]);
            acc[1][3] = wmma_bf16(a1, b3, acc[1][3]);
        }
    }

    // Epilogue. C layout: VGPR r -> row (half?8:0)+r, col = lane&15.
    int half = lane >> 4;
    int coll = lane & 15;
    #pragma unroll
    for (int i = 0; i < 2; ++i) {
        #pragma unroll
        for (int j = 0; j < 4; ++j) {
            int col = n0 + j * 16 + coll;
            float bs = bias ? bias[col] : 0.0f;
            if (bias2) bs += bias2[col];
            #pragma unroll
            for (int r = 0; r < 8; ++r) {
                int row = m0 + i * 16 + half * 8 + r;
                float v = acc[i][j][r] + bs;
                int b = row >> 10;          // / S
                int s = row & (SS - 1);
                int h = col >> 6;           // / DH
                int d = col & (DHH - 1);
                if (mode == 0) {
                    ((bf16_t*)out)[(((size_t)(b * NHH + h) * SS + s) << 6) + d] = (bf16_t)v;
                } else if (mode == 1) {
                    ((bf16_t*)out)[(((size_t)(b * NHH + h) * DHH + d) << 10) + s] = (bf16_t)v;
                } else {
                    ((float*)out)[(size_t)row * HH + col] = v;
                }
            }
        }
    }
}

// ---------------------------------------------------------------------------
// Fused flash attention (one (b,h) per blockIdx.y, 64 query rows per block,
// 4 waves, each wave owns 16 query rows; K/V streamed in 64-key tiles).
//   scores = (Q+R) K^T / 8 ; masked cols -> -1e9 ; online softmax ; O += P V
// qr,kh : [B*NH, S, DH] bf16 ;  vt : [B*NH, DH, S] bf16 ; att : [B,S,H] bf16
// ---------------------------------------------------------------------------
__global__ __launch_bounds__(128) void attn_flash(
    const bf16_t* __restrict__ qr, const bf16_t* __restrict__ kh,
    const bf16_t* __restrict__ vt, const unsigned char* __restrict__ mask,
    bf16_t* __restrict__ att)
{
    __shared__ bf16_t Plds[4][16 * 64];   // per-wave private P strip (8 KB)

    int lane = threadIdx.x & 31;
    int wave = threadIdx.x >> 5;
    int bh = blockIdx.y;
    int b  = bh >> 4;           // / NH
    int h  = bh & (NHH - 1);
    int q0 = blockIdx.x * 64 + wave * 16;

    const bf16_t* Q  = qr + (size_t)bh * SS * DHH;
    const bf16_t* Kp = kh + (size_t)bh * SS * DHH;
    const bf16_t* Vp = vt + (size_t)bh * DHH * SS;
    const unsigned char* mp = mask + (size_t)b * SS;

    int coll = lane & 15;
    int half = lane >> 4;

    // Q' fragments are loop-invariant (K-dim of the score GEMM = DH = 64)
    v16bf aq0 = load_a_frag(Q, DHH, q0, 0,  lane);
    v16bf aq1 = load_a_frag(Q, DHH, q0, 32, lane);

    v8f o[4] = {};
    float mrow[8], lrow[8];
    #pragma unroll
    for (int r = 0; r < 8; ++r) { mrow[r] = -3.0e38f; lrow[r] = 0.0f; }

    const float scl   = 0.125f;        // 1/sqrt(DH)
    const float log2e = 1.44269504f;

    for (int kb = 0; kb < SS; kb += 64) {
        // ---- scores: 4 key-tiles of 16, K-dim 64 (2 steps of 32)
        v8f sacc[4] = {};
        #pragma unroll
        for (int nt = 0; nt < 4; ++nt) {
            v16bf bk0 = load_b_frag(Kp, DHH, kb + nt * 16, 0,  lane);
            v16bf bk1 = load_b_frag(Kp, DHH, kb + nt * 16, 32, lane);
            sacc[nt] = wmma_bf16(aq0, bk0, sacc[nt]);
            sacc[nt] = wmma_bf16(aq1, bk1, sacc[nt]);
        }

        bool mk[4];
        #pragma unroll
        for (int nt = 0; nt < 4; ++nt) mk[nt] = mp[kb + nt * 16 + coll] != 0;

        // ---- online softmax per row (row stats replicated across 16 lanes)
        float alpha[8];
        #pragma unroll
        for (int r = 0; r < 8; ++r) {
            float rm = -3.0e38f;
            #pragma unroll
            for (int nt = 0; nt < 4; ++nt) {
                float sv = sacc[nt][r] * scl;
                sv = mk[nt] ? -1.0e9f : sv;
                sacc[nt][r] = sv;
                rm = fmaxf(rm, sv);
            }
            #pragma unroll
            for (int mmask = 1; mmask < 16; mmask <<= 1)
                rm = fmaxf(rm, __shfl_xor(rm, mmask, 16));
            float mnew = fmaxf(mrow[r], rm);
            alpha[r] = exp2f((mrow[r] - mnew) * log2e);
            mrow[r] = mnew;

            float rs = 0.0f;
            #pragma unroll
            for (int nt = 0; nt < 4; ++nt) {
                float p = exp2f((sacc[nt][r] - mnew) * log2e);
                Plds[wave][(half * 8 + r) * 64 + nt * 16 + coll] = (bf16_t)p;
                rs += p;
            }
            #pragma unroll
            for (int mmask = 1; mmask < 16; mmask <<= 1)
                rs += __shfl_xor(rs, mmask, 16);
            lrow[r] = lrow[r] * alpha[r] + rs;
        }

        // ---- rescale accumulator
        #pragma unroll
        for (int dt = 0; dt < 4; ++dt)
            #pragma unroll
            for (int r = 0; r < 8; ++r) o[dt][r] *= alpha[r];

        // LDS ops are in-order per wave; make the P strip visible
        asm volatile("s_wait_dscnt 0x0" ::: "memory");

        // ---- O += P * V  (A from LDS strip, B from transposed V, contiguous)
        #pragma unroll
        for (int t = 0; t < 2; ++t) {
            v16bf ap = load_a_frag(&Plds[wave][0], 64, 0, t * 32, lane);
            #pragma unroll
            for (int dt = 0; dt < 4; ++dt) {
                v16bf bv = load_b_frag(Vp, SS, dt * 16, kb + t * 32, lane);
                o[dt] = wmma_bf16(ap, bv, o[dt]);
            }
        }
        asm volatile("s_wait_dscnt 0x0" ::: "memory");
    }

    // ---- normalize and write out in [B,S,H] bf16 for the final projection
    #pragma unroll
    for (int dt = 0; dt < 4; ++dt) {
        #pragma unroll
        for (int r = 0; r < 8; ++r) {
            int srow = q0 + half * 8 + r;
            float v = o[dt][r] / lrow[r];
            att[((size_t)(b * SS + srow) * HH) + h * DHH + dt * 16 + coll] = (bf16_t)v;
        }
    }
}

// ---------------------------------------------------------------------------
// Host-side orchestration
// ---------------------------------------------------------------------------
extern "C" void kernel_launch(void* const* d_in, const int* in_sizes, int n_in,
                              void* d_out, int out_size, void* d_ws, size_t ws_size,
                              hipStream_t stream) {
    (void)in_sizes; (void)n_in; (void)out_size; (void)ws_size;

    const float* v = (const float*)d_in[0];
    const float* k = (const float*)d_in[1];
    const float* q = (const float*)d_in[2];
    const float* r = (const float*)d_in[3];
    const unsigned char* mask = (const unsigned char*)d_in[4];  // jnp.bool_ = 1 byte
    const float* Wv = (const float*)d_in[5];
    const float* bv = (const float*)d_in[6];
    const float* Wk = (const float*)d_in[7];
    const float* bk = (const float*)d_in[8];
    const float* Wq = (const float*)d_in[9];
    const float* bq = (const float*)d_in[10];
    const float* Wr = (const float*)d_in[11];
    const float* br = (const float*)d_in[12];
    const float* Wm = (const float*)d_in[13];
    const float* bm = (const float*)d_in[14];

    const size_t XN = (size_t)MM * HH;        // 8,388,608 elements
    const size_t WN = (size_t)HH * HH;        // 1,048,576 elements

    char* w = (char*)d_ws;
    bf16_t* Xv = (bf16_t*)w; w += XN * 2;
    bf16_t* Xk = (bf16_t*)w; w += XN * 2;
    bf16_t* Xq = (bf16_t*)w; w += XN * 2;
    bf16_t* Xr = (bf16_t*)w; w += XN * 2;
    bf16_t* Wvb = (bf16_t*)w; w += WN * 2;
    bf16_t* Wkb = (bf16_t*)w; w += WN * 2;
    bf16_t* Wqb = (bf16_t*)w; w += WN * 2;
    bf16_t* Wrb = (bf16_t*)w; w += WN * 2;
    bf16_t* Wmb = (bf16_t*)w; w += WN * 2;
    bf16_t* KH  = (bf16_t*)w; w += XN * 2;    // [B,NH,S,DH]
    bf16_t* QR  = (bf16_t*)w; w += XN * 2;    // [B,NH,S,DH]
    bf16_t* VT  = (bf16_t*)w; w += XN * 2;    // [B,NH,DH,S]
    bf16_t* ATT = (bf16_t*)w; w += XN * 2;    // [B,S,H]

    // fp32 -> bf16
    cvt_f32_bf16<<<4096, 256, 0, stream>>>(v, Xv, (int)XN);
    cvt_f32_bf16<<<4096, 256, 0, stream>>>(k, Xk, (int)XN);
    cvt_f32_bf16<<<4096, 256, 0, stream>>>(q, Xq, (int)XN);
    cvt_f32_bf16<<<4096, 256, 0, stream>>>(r, Xr, (int)XN);
    cvt_f32_bf16<<<1024, 256, 0, stream>>>(Wv, Wvb, (int)WN);
    cvt_f32_bf16<<<1024, 256, 0, stream>>>(Wk, Wkb, (int)WN);
    cvt_f32_bf16<<<1024, 256, 0, stream>>>(Wq, Wqb, (int)WN);
    cvt_f32_bf16<<<1024, 256, 0, stream>>>(Wr, Wrb, (int)WN);
    cvt_f32_bf16<<<1024, 256, 0, stream>>>(Wm, Wmb, (int)WN);

    dim3 ggrid(HH / 128, MM / 128);   // (8, 64)
    // V projection -> transposed head layout
    gemm_bf16<<<ggrid, 256, 0, stream>>>(Xv, nullptr, Wvb, nullptr, bv, nullptr, VT, 1);
    // K projection -> head layout
    gemm_bf16<<<ggrid, 256, 0, stream>>>(Xk, nullptr, Wkb, nullptr, bk, nullptr, KH, 0);
    // fused (Q+R) projection -> head layout
    gemm_bf16<<<ggrid, 256, 0, stream>>>(Xq, Xr, Wqb, Wrb, bq, br, QR, 0);

    // fused flash attention
    dim3 agrid(SS / 64, BB * NHH);   // (16, 128)
    attn_flash<<<agrid, 128, 0, stream>>>(QR, KH, VT, mask, ATT);

    // output projection -> fp32 d_out
    gemm_bf16<<<ggrid, 256, 0, stream>>>(ATT, nullptr, Wmb, nullptr, bm, nullptr, d_out, 2);
}
